// CudaFastWeightLinearTransformerLayer_15891378995473
// MI455X (gfx1250) — matrix operand order repaired and checked by reading, not compiled, not run on hardware
//
#include <hip/hip_runtime.h>
#include <hip/hip_bf16.h>
#include <math.h>

// Problem constants (from reference)
#define SLEN    1024
#define BSZ     8
#define D_MODEL 1024
#define N_HEAD  16
#define D_HEAD  64
#define HEAD_STRIDE (3 * D_HEAD + 1)            // 193
#define QKVB_COLS   (N_HEAD * HEAD_STRIDE)      // 3088
#define TOKENS      (SLEN * BSZ)                // 8192
#define EPS_LN  1e-5f
#define SCALE_O 0.125f                          // 1/sqrt(64)

typedef __bf16 bf16;
typedef bf16  v16bf __attribute__((ext_vector_type(16)));
typedef float v8f   __attribute__((ext_vector_type(8)));

// ---------------------------------------------------------------------------
// GEMM config: block tile 128x128, BK=32, 8 waves as 4(M) x 2(N),
// wave tile 32x64 -> 8 x v_wmma_f32_16x16x32_bf16 per K-step per wave.
// fp32 tiles are DMA'd into double-buffered LDS with
// GLOBAL_LOAD_ASYNC_TO_LDS_B128 (ASYNCcnt), converted to bf16 at
// fragment-build time (v_cvt_pk_bf16_f32 co-executes with the matrix pipe).
// All 6 fragment octets (a0,a1,b0..b3) are materialized before the WMMA
// burst so their live ranges overlap -> distinct registers -> the
// WMMA->VALU WAR hazard slots are filled with real work, not v_nops.
// ---------------------------------------------------------------------------
#define BM 128
#define BN 128
#define BK 32
#define LDA_F 36    // fp32 elements; 36*4 bytes = multiple of 16 (b128 dst align)
#define LDB_F 132   // fp32 elements; 132*4 bytes = multiple of 16

__device__ __forceinline__ void async_copy_b128(unsigned lds_off,
                                                const float* gptr) {
  asm volatile("global_load_async_to_lds_b128 %0, %1, off"
               :: "v"(lds_off), "v"(gptr)
               : "memory");
}

__device__ __forceinline__ unsigned lds_addr_of(const float* p) {
  return (unsigned)(uintptr_t)p;   // low 32 bits of flat LDS ptr = LDS offset
}

__global__ __launch_bounds__(256)
void gemm_bf16_wmma(const float* __restrict__ A, const float* __restrict__ B,
                    const float* __restrict__ resid, float* __restrict__ C,
                    int M, int N, int K) {
  __shared__ float Abuf[2][BM * LDA_F];   // ~36.9 KB
  __shared__ float Bbuf[2][BK * LDB_F];   // ~33.8 KB

  const int tid  = threadIdx.x;
  const int lane = tid & 31;
  const int wave = tid >> 5;        // 0..7
  const int wm   = wave & 3;        // 32-row stripe within 128
  const int wn   = wave >> 2;       // 0..1 -> 64-col half within 128
  const int half = lane >> 4;       // lane group (ISA 16-bit fragment layout)
  const int lm   = lane & 15;

  const int blockM = blockIdx.y * BM;
  const int blockN = blockIdx.x * BN;
  const bool edge  = (blockN + BN > N);   // uniform across the block

  v8f acc0[4] = {};
  v8f acc1[4] = {};

  // ---- async-stage one BKxBM / BKxBN tile pair into LDS buffer `sel` ----
  auto issue_tile = [&](int k0, int sel) {
    float* Ab = Abuf[sel];
    float* Bb = Bbuf[sel];
    // A tile: 128 rows x 32 cols fp32 = 1024 float4 slots, 4 per thread
#pragma unroll
    for (int i = 0; i < 4; ++i) {
      int idx = tid + i * 256;
      int row = idx >> 3;
      int c4  = (idx & 7) << 2;
      const float* g = A + (size_t)(blockM + row) * K + k0 + c4;
      async_copy_b128(lds_addr_of(Ab + row * LDA_F + c4), g);
    }
    // B tile: 32 rows x 128 cols fp32, row-major in LDS
#pragma unroll
    for (int i = 0; i < 4; ++i) {
      int idx = tid + i * 256;
      int kr  = idx >> 5;
      int c4  = (idx & 31) << 2;
      int gc  = blockN + c4;
      const float* g = B + (size_t)(k0 + kr) * N + gc;
      float* dst = Bb + kr * LDB_F + c4;
      if (gc + 3 < N) {
        async_copy_b128(lds_addr_of(dst), g);
      } else {                       // rare edge tile: guarded scalar staging
        dst[0] = (gc + 0 < N) ? g[0] : 0.f;
        dst[1] = (gc + 1 < N) ? g[1] : 0.f;
        dst[2] = (gc + 2 < N) ? g[2] : 0.f;
        dst[3] = 0.f;
      }
    }
  };

  const int nk = K / BK;
  issue_tile(0, 0);

  for (int kt = 0; kt < nk; ++kt) {
    const int sel = kt & 1;
    const bool has_next = (kt + 1 < nk);
    if (has_next) issue_tile((kt + 1) * BK, sel ^ 1);

    // Fence: current tile's 8 async loads complete (next tile's 8 may remain
    // in flight). Edge blocks use a conservative full drain.
    if (has_next && !edge)
      asm volatile("s_wait_asynccnt 0x8" ::: "memory");
    else
      asm volatile("s_wait_asynccnt 0x0" ::: "memory");
    __syncthreads();

    const float* Ab = Abuf[sel];
    const float* Bb = Bbuf[sel];

    // A fragments: 16x32 bf16; elems 0-7 -> K=half*8+e, elems 8-15 ->
    // K=16+half*8+(e-8); row M = lane%16.
    v16bf a0, a1;
    {
      const float* Ar0 = Ab + (wm * 32 + lm) * LDA_F;
      const float* Ar1 = Ar0 + 16 * LDA_F;
#pragma unroll
      for (int g = 0; g < 2; ++g)
#pragma unroll
        for (int r = 0; r < 4; ++r) {
          int kk = g * 16 + half * 8 + 2 * r;
          a0[g * 8 + 2 * r]     = (bf16)Ar0[kk];
          a0[g * 8 + 2 * r + 1] = (bf16)Ar0[kk + 1];
          a1[g * 8 + 2 * r]     = (bf16)Ar1[kk];
          a1[g * 8 + 2 * r + 1] = (bf16)Ar1[kk + 1];
        }
    }
    // B fragments: 32x16 bf16; col N = lane%16, K = (e/8)*16 + half*8 + e%8.
    // All four fragments live simultaneously -> 4 distinct register octets.
    v16bf bf4[4];
#pragma unroll
    for (int tn = 0; tn < 4; ++tn) {
      const float* Bc = Bb + wn * 64 + tn * 16 + lm;
#pragma unroll
      for (int e = 0; e < 16; ++e) bf4[tn][e] = (bf16)Bc[((e >> 3) * 16 + half * 8 + (e & 7)) * LDB_F];
    }
    // WMMA burst: 8 independent matrix ops (each reads/writes only its own
    // accumulator; fragment sources are read-only -> no intra-burst hazards).
#pragma unroll
    for (int tn = 0; tn < 4; ++tn) {
      acc0[tn] = __builtin_amdgcn_wmma_f32_16x16x32_bf16(
          false, a0, false, bf4[tn], (short)0, acc0[tn], false, false);
      acc1[tn] = __builtin_amdgcn_wmma_f32_16x16x32_bf16(
          false, a1, false, bf4[tn], (short)0, acc1[tn], false, false);
    }
    __syncthreads();   // protect buffer reuse two iterations ahead
  }

  // ---- store (C/D layout: vgpr r -> row half*8+r, col = lane%16) ----
#pragma unroll
  for (int tn = 0; tn < 4; ++tn) {
    int col = blockN + wn * 64 + tn * 16 + lm;
    if (col >= N) continue;
#pragma unroll
    for (int i = 0; i < 2; ++i) {
      const v8f& acc = i ? acc1[tn] : acc0[tn];
#pragma unroll
      for (int r = 0; r < 8; ++r) {
        int row = blockM + wm * 32 + i * 16 + half * 8 + r;
        if (row < M) {
          size_t off = (size_t)row * N + col;
          float v = acc[r];
          if (resid) v += resid[off];
          C[off] = v;
        }
      }
    }
  }
}

// ---------------------------------------------------------------------------
// In-place activation on qkvb: q,k -> (elu+1)/sum, beta -> sigmoid.
// One wave per (token, head); lane covers elements e and e+32 of D_HEAD=64.
// ---------------------------------------------------------------------------
__global__ __launch_bounds__(256)
void qkvb_activate(float* __restrict__ qkvb) {
  int gid  = blockIdx.x * 8 + (threadIdx.x >> 5);   // token*16 + head
  int lane = threadIdx.x & 31;
  float* base = qkvb + (size_t)(gid >> 4) * QKVB_COLS + (gid & 15) * HEAD_STRIDE;

#pragma unroll
  for (int part = 0; part < 2; ++part) {            // 0: q, 1: k
    float x0 = base[part * 64 + lane];
    float x1 = base[part * 64 + lane + 32];
    float p0 = x0 > 0.f ? x0 + 1.f : __expf(x0);    // elu(x)+1
    float p1 = x1 > 0.f ? x1 + 1.f : __expf(x1);
    float s = p0 + p1;
#pragma unroll
    for (int off = 16; off >= 1; off >>= 1) s += __shfl_xor(s, off, 32);
    float inv = 1.f / s;
    base[part * 64 + lane]      = p0 * inv;
    base[part * 64 + lane + 32] = p1 * inv;
  }
  if (lane == 0) {
    float b = base[192];
    base[192] = 1.f / (1.f + __expf(-b));           // sigmoid
  }
}

// ---------------------------------------------------------------------------
// Sequential delta-rule fast-weight scan. One 64-thread block per (b, head).
// Thread t keeps row t of the 64x64 state W in VGPRs; q/k/v broadcast via LDS.
// global_prefetch pulls the t+2 step ahead of the dependent chain.
// ---------------------------------------------------------------------------
__global__ __launch_bounds__(64)
void fast_weight_scan(const float* __restrict__ qkvb, float* __restrict__ outs) {
  __shared__ float qls[64], kls[64], vls[64];
  __shared__ float betas;
  const int b    = blockIdx.x >> 4;
  const int head = blockIdx.x & 15;
  const int tid  = threadIdx.x;

  float Wrow[64];
#pragma unroll
  for (int j = 0; j < 64; ++j) Wrow[j] = 0.f;

  for (int t = 0; t < SLEN; ++t) {
    const int token = t * BSZ + b;
    const float* base = qkvb + (size_t)token * QKVB_COLS + head * HEAD_STRIDE;
    qls[tid] = base[tid];
    kls[tid] = base[64 + tid];
    vls[tid] = base[128 + tid];
    if (tid == 0) betas = base[192];
    if (t + 2 < SLEN) {
      const float* pf = qkvb + (size_t)((t + 2) * BSZ + b) * QKVB_COLS
                        + head * HEAD_STRIDE;
      __builtin_prefetch(pf + tid, 0, 0);       // q/k halves
      __builtin_prefetch(pf + 128 + tid, 0, 0); // v + beta
    }
    __syncthreads();

    float vold = 0.f;
#pragma unroll 8
    for (int j = 0; j < 64; ++j) vold += Wrow[j] * kls[j];
    float delta = betas * (vls[tid] - vold);
    float out = 0.f;
#pragma unroll 8
    for (int j = 0; j < 64; ++j) {
      Wrow[j] += delta * kls[j];
      out += Wrow[j] * qls[j];
    }
    outs[((size_t)token * N_HEAD + head) * D_HEAD + tid] = SCALE_O * out;
    __syncthreads();
  }
}

// ---------------------------------------------------------------------------
// LayerNorm over D_MODEL=1024; one 256-thread block per row.
// ---------------------------------------------------------------------------
__global__ __launch_bounds__(256)
void layernorm_kernel(const float* __restrict__ x, const float* __restrict__ g,
                      const float* __restrict__ bta, float* __restrict__ out) {
  __shared__ float red[16];
  const int row = blockIdx.x;
  const float* xr = x + (size_t)row * D_MODEL;
  float lv[4];
  float s = 0.f, ss = 0.f;
#pragma unroll
  for (int i = 0; i < 4; ++i) {
    float v = xr[threadIdx.x + i * 256];
    lv[i] = v;
    s += v; ss += v * v;
  }
#pragma unroll
  for (int off = 16; off >= 1; off >>= 1) {
    s  += __shfl_xor(s, off, 32);
    ss += __shfl_xor(ss, off, 32);
  }
  int wave = threadIdx.x >> 5, lane = threadIdx.x & 31;
  if (lane == 0) { red[wave] = s; red[8 + wave] = ss; }
  __syncthreads();
  if (wave == 0) {
    float a = lane < 8 ? red[lane] : 0.f;
    float b = lane < 8 ? red[8 + lane] : 0.f;
#pragma unroll
    for (int off = 4; off >= 1; off >>= 1) {
      a += __shfl_xor(a, off, 32);
      b += __shfl_xor(b, off, 32);
    }
    if (lane == 0) { red[0] = a; red[1] = b; }
  }
  __syncthreads();
  float mu   = red[0] * (1.f / D_MODEL);
  float var  = red[1] * (1.f / D_MODEL) - mu * mu;
  float rstd = rsqrtf(var + EPS_LN);
#pragma unroll
  for (int i = 0; i < 4; ++i) {
    int c = threadIdx.x + i * 256;
    out[(size_t)row * D_MODEL + c] = (lv[i] - mu) * rstd * g[c] + bta[c];
  }
}

// ---------------------------------------------------------------------------
extern "C" void kernel_launch(void* const* d_in, const int* in_sizes, int n_in,
                              void* d_out, int out_size, void* d_ws, size_t ws_size,
                              hipStream_t stream) {
  const float* h      = (const float*)d_in[0];
  const float* W_qkvb = (const float*)d_in[1];
  const float* W_o    = (const float*)d_in[2];
  const float* ln_g   = (const float*)d_in[3];
  const float* ln_b   = (const float*)d_in[4];
  float* out = (float*)d_out;

  // Workspace layout (fp32):
  //   qkvb  : TOKENS x 3088  (activated in place)
  //   outs  : TOKENS x 1024  (scaled fast-weight outputs = layer_out)
  //   x     : TOKENS x 1024  (h + attn_out, pre-LayerNorm)
  float* qkvb = (float*)d_ws;
  float* outs = qkvb + (size_t)TOKENS * QKVB_COLS;
  float* x    = outs + (size_t)TOKENS * D_MODEL;

  // 1) qkvb = h @ W_qkvb
  dim3 g1((QKVB_COLS + BN - 1) / BN, (TOKENS + BM - 1) / BM);
  gemm_bf16_wmma<<<g1, 256, 0, stream>>>(h, W_qkvb, nullptr, qkvb,
                                         TOKENS, QKVB_COLS, D_MODEL);
  // 2) activations in place
  qkvb_activate<<<(TOKENS * N_HEAD) / 8, 256, 0, stream>>>(qkvb);
  // 3) sequential fast-weight scan
  fast_weight_scan<<<BSZ * N_HEAD, 64, 0, stream>>>(qkvb, outs);
  // 4) x = h + outs @ W_o
  dim3 g2((D_MODEL + BN - 1) / BN, (TOKENS + BM - 1) / BM);
  gemm_bf16_wmma<<<g2, 256, 0, stream>>>(outs, W_o, h, x,
                                         TOKENS, D_MODEL, D_MODEL);
  // 5) LayerNorm -> out
  layernorm_kernel<<<TOKENS, 256, 0, stream>>>(x, ln_g, ln_b, out);
}
